// MultiheadAttention_53772990545996
// MI455X (gfx1250) — matrix operand order repaired and checked by reading, not compiled, not run on hardware
//
#include <hip/hip_runtime.h>
#include <hip/hip_bf16.h>

// ---------------------------------------------------------------------------
// MultiheadAttention (xPos) forward for MI455X / gfx1250, wave32 + WMMA.
// B=2 T=2048 E=1024 H=16 D=64.  Memory-bound: rel_pos read + probs write are
// 537 MB each -> fuse scores+softmax+PV using 296 KB of LDS per workgroup so
// probs are written exactly once and never re-read from HBM.
// ---------------------------------------------------------------------------

#define B_  2
#define T_  2048
#define E_  1024
#define H_  16
#define D_  64
#define BT_ (B_*T_)
#define BH_ (B_*H_)

typedef __attribute__((ext_vector_type(16))) __bf16 v16bf;
typedef __attribute__((ext_vector_type(8)))  float  v8f;

#define KP32 40   // padded LDS row stride (bf16) for K-step=32 tiles
#define KP64 72   // padded LDS row stride (bf16) for K=64 tiles

union V16U { v16bf bf; uint4 u[2]; };

// A-fragment (16x32 bf16, §7.12.2): lane h8=lane>>4 holds k in
// [h8*8, h8*8+8) and [16+h8*8, 16+h8*8+8) for row m = lane&15.
__device__ inline v16bf frag_a(const __bf16* row, int h8) {
  V16U r;
  r.u[0] = *(const uint4*)(row + h8*8);
  r.u[1] = *(const uint4*)(row + 16 + h8*8);
  return r.bf;
}
// B-fragment (32x16 bf16): lane holds col n = lane&15, k in
// [h8*8, h8*8+8) and [16+h8*8, 16+h8*8+8).
__device__ inline v16bf frag_b(const __bf16* row, int h8) {
  V16U r;
  r.u[0] = *(const uint4*)(row + h8*8);
  r.u[1] = *(const uint4*)(row + 16 + h8*8);
  return r.bf;
}
// A-fragment sourced from f32 LDS (probs) with on-the-fly bf16 convert.
__device__ inline v16bf frag_a_f32(const float* row, int h8, int kbase) {
  const float4* p0 = (const float4*)(row + kbase + h8*8);
  const float4* p1 = (const float4*)(row + kbase + 16 + h8*8);
  float4 a0 = p0[0], a1 = p0[1], b0 = p1[0], b1 = p1[1];
  v16bf r;
  r[0]=(__bf16)a0.x; r[1]=(__bf16)a0.y; r[2]=(__bf16)a0.z; r[3]=(__bf16)a0.w;
  r[4]=(__bf16)a1.x; r[5]=(__bf16)a1.y; r[6]=(__bf16)a1.z; r[7]=(__bf16)a1.w;
  r[8]=(__bf16)b0.x; r[9]=(__bf16)b0.y; r[10]=(__bf16)b0.z; r[11]=(__bf16)b0.w;
  r[12]=(__bf16)b1.x; r[13]=(__bf16)b1.y; r[14]=(__bf16)b1.z; r[15]=(__bf16)b1.w;
  return r;
}

__device__ inline v8f wmma_bf16(v16bf a, v16bf b, v8f c) {
  return __builtin_amdgcn_wmma_f32_16x16x32_bf16(false, a, false, b,
                                                 (short)0, c, false, false);
}

// ---------------------------------------------------------------------------
__global__ void cvt_bf16_kernel(const float* __restrict__ src,
                                __bf16* __restrict__ dst, int n) {
  int i = (blockIdx.x * 256 + threadIdx.x) * 4;
  if (i >= n) return;
  float4 v = *(const float4*)(src + i);
  dst[i+0] = (__bf16)v.x; dst[i+1] = (__bf16)v.y;
  dst[i+2] = (__bf16)v.z; dst[i+3] = (__bf16)v.w;
}

// ---------------------------------------------------------------------------
// C[m,n] = (sum_k A[m,k]*W[n,k] + bias[n]) * scale    (A,W bf16; C f32)
// Block 256 thr = 8 waves as 2x4; block tile 32x256; K-step 32.
__global__ __launch_bounds__(256)
void gemm_bf16_kernel(const __bf16* __restrict__ A, const __bf16* __restrict__ Bw,
                      const float* __restrict__ bias, float* __restrict__ Cout,
                      int M, int N, int K, float scale)
{
  __shared__ __bf16 lA[32 * KP32];
  __shared__ __bf16 lB[256 * KP32];
  const int tid = threadIdx.x;
  const int wave = tid >> 5, lane = tid & 31;
  const int h8 = lane >> 4, l15 = lane & 15;
  const int wm = (wave >> 2) * 16, wn = (wave & 3) * 64;
  const int m0 = blockIdx.x * 32, n0 = blockIdx.y * 256;

  v8f acc[4] = {};
  for (int k0 = 0; k0 < K; k0 += 32) {
    __syncthreads();
    { int r = tid >> 3, c = (tid & 7) * 4;
      *(uint2*)&lA[r*KP32 + c] = *(const uint2*)&A[(size_t)(m0 + r)*K + k0 + c]; }
    { const uint4* src = (const uint4*)&Bw[(size_t)(n0 + tid)*K + k0];
      uint4* dst = (uint4*)&lB[tid*KP32];
      dst[0]=src[0]; dst[1]=src[1]; dst[2]=src[2]; dst[3]=src[3]; }
    __syncthreads();
    v16bf af = frag_a(&lA[(wm + l15)*KP32], h8);
#pragma unroll
    for (int t = 0; t < 4; ++t) {
      v16bf bfr = frag_b(&lB[(wn + t*16 + l15)*KP32], h8);
      acc[t] = wmma_bf16(af, bfr, acc[t]);
    }
  }
#pragma unroll
  for (int t = 0; t < 4; ++t) {
    int n = n0 + wn + t*16 + l15;
    float bb = bias[n];
#pragma unroll
    for (int r = 0; r < 8; ++r) {
      int m = m0 + wm + r + 8*h8;
      Cout[(size_t)m*N + n] = (acc[t][r] + bb) * scale;
    }
  }
}

// ---------------------------------------------------------------------------
// xPos rotary + per-head bf16 pack.  One thread per (b,h,t, d-pair).
__global__ void xpos_pack_kernel(const float* __restrict__ src,
                                 __bf16* __restrict__ dst, int downscale)
{
  int gid = blockIdx.x * 256 + threadIdx.x;
  int j  = gid & 31;                 // pair index (D/2 = 32)
  int t  = (gid >> 5) & (T_ - 1);
  int bh = gid >> 16;                // / (32*2048)
  int b = bh / H_, h = bh % H_;
  const float* x = src + (size_t)(b*T_ + t)*E_ + h*D_ + 2*j;
  float x1 = x[0], x2 = x[1];
  float base = (2.f*j + 0.4f*D_) / (1.4f*D_);
  float pos  = (float)t - (float)(T_/2);
  float sc   = __powf(base, pos * (1.f/512.f));
  if (downscale) sc = 1.f / sc;
  float invf = __powf(10000.f, -(float)j * (1.f/32.f));
  float ang  = (float)t * invf;
  float s, c; __sincosf(ang, &s, &c);
  c *= sc; s *= sc;
  __bf16* d = dst + (size_t)(bh*T_ + t)*D_ + 2*j;
  d[0] = (__bf16)(x1*c - x2*s);
  d[1] = (__bf16)(x2*c + x1*s);
}

// v^T pack: vT[b,h,d,s] = v[b,s,h*D+d] (bf16), coalesced along s.
__global__ void pack_vt_kernel(const float* __restrict__ vf,
                               __bf16* __restrict__ vT)
{
  int gid = blockIdx.x * 256 + threadIdx.x;      // over B*H*D*T
  int s  = gid & (T_ - 1);
  int d  = (gid >> 11) & (D_ - 1);
  int bh = gid >> 17;                            // / (D*T)
  int b = bh / H_, h = bh % H_;
  vT[(size_t)gid] = (__bf16)vf[(size_t)(b*T_ + s)*E_ + h*D_ + d];
}

// ---------------------------------------------------------------------------
// Fused attention: per block = 32 query rows of one (b,h).
//   phase 1: S = q@k^T + mask + rel_pos  -> 32x2048 f32 stripe in LDS (256 KB)
//   phase 2: softmax rows in LDS, single probs write to HBM
//   phase 3: PV via WMMA reading probs from LDS (never re-read from HBM)
// Dynamic LDS: 262144 (S) + 36864 (k/v tile) + 4608 (q tile) = 303616 B
// (fits CDNA5's 320 KB/WGP; one workgroup per WGP).
__global__ __launch_bounds__(256)
void attn_fused_kernel(const __bf16* __restrict__ qh, const __bf16* __restrict__ kh,
                       const __bf16* __restrict__ vT, const float* __restrict__ mask,
                       const float* __restrict__ rel, float* __restrict__ probs,
                       float* __restrict__ attnf)
{
  extern __shared__ char smem[];
  float*  S  = (float*)smem;                                       // 32 x 2048
  __bf16* kv = (__bf16*)(smem + 32*T_*sizeof(float));              // 256x72 / 64x40
  __bf16* qs = (__bf16*)(smem + 32*T_*sizeof(float) + 256*KP64*2); // 32x72

  const int tid  = threadIdx.x;
  const int wave = tid >> 5, lane = tid & 31;
  const int h8 = lane >> 4, l15 = lane & 15;
  const int t0 = blockIdx.x * 32;
  const int bh = blockIdx.y;
  const int b = bh / H_, h = bh % H_;

  // stage q tile (32 x 64 bf16)
  { int r = tid >> 3, c = (tid & 7) * 8;
    *(uint4*)&qs[r*KP64 + c] = *(const uint4*)&qh[((size_t)bh*T_ + t0 + r)*D_ + c]; }
  __syncthreads();

  const int wm = (wave >> 2) * 16, wn = (wave & 3) * 64;
  v16bf aq0 = frag_a(&qs[(wm + l15)*KP64], h8);        // k 0..31
  v16bf aq1 = frag_a(&qs[(wm + l15)*KP64 + 32], h8);   // k 32..63

  // ---- phase 1: scores ----
  for (int n0 = 0; n0 < T_; n0 += 256) {
    __syncthreads();
    { const uint4* src = (const uint4*)&kh[((size_t)bh*T_ + n0 + tid)*D_];
      uint4* dst = (uint4*)&kv[tid*KP64];
#pragma unroll
      for (int i = 0; i < 8; ++i) dst[i] = src[i]; }
    if (n0 + 256 < T_)
      __builtin_prefetch(&kh[((size_t)bh*T_ + n0 + 256 + tid)*D_], 0, 1);
    __syncthreads();
#pragma unroll
    for (int t = 0; t < 4; ++t) {
      const __bf16* krow = &kv[(wn + t*16 + l15)*KP64];
      v8f acc = {};
      acc = wmma_bf16(aq0, frag_b(krow, h8), acc);
      acc = wmma_bf16(aq1, frag_b(krow + 32, h8), acc);
      int n = n0 + wn + t*16 + l15;
#pragma unroll
      for (int r = 0; r < 8; ++r) {
        int mrow = wm + r + 8*h8;
        int tq = t0 + mrow;
        S[mrow*T_ + n] = acc[r] + mask[(size_t)tq*T_ + n]
                       + rel[((size_t)bh*T_ + tq)*T_ + n];
      }
    }
  }
  __syncthreads();

  // ---- phase 2: softmax per row (wave handles 4 rows) + probs write ----
#pragma unroll
  for (int rr = 0; rr < 4; ++rr) {
    int row = wave*4 + rr;
    float* sr = S + row*T_;
    float mx = -3.4e38f;
    for (int c = lane; c < T_; c += 32) mx = fmaxf(mx, sr[c]);
#pragma unroll
    for (int off = 16; off; off >>= 1) mx = fmaxf(mx, __shfl_xor(mx, off, 32));
    float sum = 0.f;
    for (int c = lane; c < T_; c += 32) {
      float e = __expf(sr[c] - mx); sr[c] = e; sum += e;
    }
#pragma unroll
    for (int off = 16; off; off >>= 1) sum += __shfl_xor(sum, off, 32);
    float inv = 1.f / sum;
    float* pg = probs + (((size_t)h*B_ + b)*T_ + t0 + row)*T_;  // (H,B,T,T)
    for (int c = lane; c < T_; c += 32) {
      float p = sr[c] * inv; sr[c] = p; pg[c] = p;
    }
  }
  __syncthreads();

  // ---- phase 3: PV.  Wave (wr,wc) owns one 16x16 tile of the 32x64 result.
  const int wr = wave >> 2, wc = wave & 3;
  v8f accp = {};
  for (int s0 = 0; s0 < T_; s0 += 32) {
    __syncthreads();
    if (tid < D_) {
      const uint4* src = (const uint4*)&vT[((size_t)bh*D_ + tid)*T_ + s0];
      uint4* dst = (uint4*)&kv[tid*KP32];
      dst[0]=src[0]; dst[1]=src[1]; dst[2]=src[2]; dst[3]=src[3];
    }
    __syncthreads();
    v16bf ap = frag_a_f32(&S[(wr*16 + l15)*T_], h8, s0);
    v16bf bp = frag_b(&kv[(wc*16 + l15)*KP32], h8);
    accp = wmma_bf16(ap, bp, accp);
  }
#pragma unroll
  for (int r = 0; r < 8; ++r) {
    int trow = t0 + wr*16 + r + 8*h8;
    attnf[((size_t)b*T_ + trow)*E_ + h*D_ + wc*16 + l15] = accp[r];
  }
}

// ---------------------------------------------------------------------------
__global__ __launch_bounds__(256)
void layernorm_kernel(const float* __restrict__ x, const float* __restrict__ g,
                      const float* __restrict__ bb, __bf16* __restrict__ y)
{
  __shared__ float red[256];
  int row = blockIdx.x, tid = threadIdx.x;
  const float* xr = x + (size_t)row*E_;
  float4 v = *(const float4*)(xr + tid*4);
  red[tid] = v.x + v.y + v.z + v.w;
  __syncthreads();
  for (int o = 128; o; o >>= 1) { if (tid < o) red[tid] += red[tid+o]; __syncthreads(); }
  float mu = red[0] * (1.f/E_);
  __syncthreads();
  float dx = v.x-mu, dy = v.y-mu, dz = v.z-mu, dw = v.w-mu;
  red[tid] = dx*dx + dy*dy + dz*dz + dw*dw;
  __syncthreads();
  for (int o = 128; o; o >>= 1) { if (tid < o) red[tid] += red[tid+o]; __syncthreads(); }
  float rstd = rsqrtf(red[0] * (1.f/E_) + 1e-5f);
  int e = tid*4;
  float4 gg = *(const float4*)(g + e);
  float4 bv = *(const float4*)(bb + e);
  __bf16* yr = y + (size_t)row*E_ + e;
  yr[0] = (__bf16)(dx*rstd*gg.x + bv.x);
  yr[1] = (__bf16)(dy*rstd*gg.y + bv.y);
  yr[2] = (__bf16)(dz*rstd*gg.z + bv.z);
  yr[3] = (__bf16)(dw*rstd*gg.w + bv.w);
}

// ---------------------------------------------------------------------------
extern "C" void kernel_launch(void* const* d_in, const int* in_sizes, int n_in,
                              void* d_out, int out_size, void* d_ws, size_t ws_size,
                              hipStream_t stream)
{
  (void)in_sizes; (void)n_in; (void)out_size; (void)ws_size;
  const float* query = (const float*)d_in[0];
  const float* key   = (const float*)d_in[1];
  const float* value = (const float*)d_in[2];
  const float* mask  = (const float*)d_in[3];
  const float* rel   = (const float*)d_in[4];
  const float* Wq = (const float*)d_in[5];  const float* bq = (const float*)d_in[6];
  const float* Wk = (const float*)d_in[7];  const float* bk = (const float*)d_in[8];
  const float* Wv = (const float*)d_in[9];  const float* bv = (const float*)d_in[10];
  const float* Wo = (const float*)d_in[11]; const float* bo = (const float*)d_in[12];
  const float* ln_g = (const float*)d_in[13];
  const float* ln_b = (const float*)d_in[14];

  float* out   = (float*)d_out;                       // (B,T,E)
  float* probs = out + (size_t)BT_ * E_;              // (H,B,T,T)

  // workspace carve-out (~112 MB)
  char* ws = (char*)d_ws; size_t off = 0;
  auto carve = [&](size_t bytes) {
    void* p = ws + off; off = (off + bytes + 255) & ~(size_t)255; return p;
  };
  __bf16* wqb = (__bf16*)carve((size_t)E_*E_*2);
  __bf16* wkb = (__bf16*)carve((size_t)E_*E_*2);
  __bf16* wvb = (__bf16*)carve((size_t)E_*E_*2);
  __bf16* wob = (__bf16*)carve((size_t)E_*E_*2);
  __bf16* xb  = (__bf16*)carve((size_t)BT_*E_*2);     // reused for q/k/v inputs
  float*  qf  = (float*) carve((size_t)BT_*E_*4);
  float*  kf  = (float*) carve((size_t)BT_*E_*4);
  float*  vf  = (float*) carve((size_t)BT_*E_*4);
  __bf16* qhb = (__bf16*)carve((size_t)BH_*T_*D_*2);
  __bf16* khb = (__bf16*)carve((size_t)BH_*T_*D_*2);
  __bf16* vtb = (__bf16*)carve((size_t)BH_*D_*T_*2);
  float*  atf = (float*) carve((size_t)BT_*E_*4);
  __bf16* xo  = (__bf16*)carve((size_t)BT_*E_*2);

  const int nW  = E_*E_;     // 1,048,576
  const int nX  = BT_*E_;    // 4,194,304
  const dim3 gemmGrid(BT_/32, E_/256);
  const float qscale = 0.125f;  // D^-0.5

  // weights -> bf16
  cvt_bf16_kernel<<<nW/1024, 256, 0, stream>>>(Wq, wqb, nW);
  cvt_bf16_kernel<<<nW/1024, 256, 0, stream>>>(Wk, wkb, nW);
  cvt_bf16_kernel<<<nW/1024, 256, 0, stream>>>(Wv, wvb, nW);
  cvt_bf16_kernel<<<nW/1024, 256, 0, stream>>>(Wo, wob, nW);

  // QKV projections (xb reused serially on the stream)
  cvt_bf16_kernel<<<nX/1024, 256, 0, stream>>>(query, xb, nX);
  gemm_bf16_kernel<<<gemmGrid, 256, 0, stream>>>(xb, wqb, bq, qf, BT_, E_, E_, qscale);
  cvt_bf16_kernel<<<nX/1024, 256, 0, stream>>>(key, xb, nX);
  gemm_bf16_kernel<<<gemmGrid, 256, 0, stream>>>(xb, wkb, bk, kf, BT_, E_, E_, 1.0f);
  cvt_bf16_kernel<<<nX/1024, 256, 0, stream>>>(value, xb, nX);
  gemm_bf16_kernel<<<gemmGrid, 256, 0, stream>>>(xb, wvb, bv, vf, BT_, E_, E_, 1.0f);

  // xPos + per-head packing
  xpos_pack_kernel<<<(BH_*T_*32)/256, 256, 0, stream>>>(qf, qhb, 0);
  xpos_pack_kernel<<<(BH_*T_*32)/256, 256, 0, stream>>>(kf, khb, 1);
  pack_vt_kernel<<<(BH_*D_*T_)/256, 256, 0, stream>>>(vf, vtb);

  // fused attention: 296 KB dynamic LDS per workgroup (CDNA5: 320 KB/WGP)
  const size_t smemBytes = 32*T_*sizeof(float) + 256*KP64*2 + 32*KP64*2;
  (void)hipFuncSetAttribute((const void*)attn_fused_kernel,
                            hipFuncAttributeMaxDynamicSharedMemorySize,
                            (int)smemBytes);
  attn_fused_kernel<<<dim3(T_/32, BH_), 256, smemBytes, stream>>>(
      qhb, khb, vtb, mask, rel, probs, atf);

  // LayerNorm -> bf16, then output projection
  layernorm_kernel<<<BT_, 256, 0, stream>>>(atf, ln_g, ln_b, xo);
  gemm_bf16_kernel<<<gemmGrid, 256, 0, stream>>>(xo, wob, bo, out, BT_, E_, E_, 1.0f);
}
